// FrequencySparseFFTConv_52767968199449
// MI455X (gfx1250) — matrix operand order repaired and checked by reading, not compiled, not run on hardware
//
#include <hip/hip_runtime.h>
#include <hip/hip_bf16.h>

// ---------------------------------------------------------------------------
// Frequency-sparse FFT conv as 3 GEMM stages on CDNA5 WMMA (bf16x2 split),
// with the DFT basis tiles streamed into LDS by the Tensor Data Mover.
//   Stage T : build cos/sin DFT basis tables (bf16 hi/lo) in workspace.
//   Stage 1 : Xc/Xs = A @ [cos|sin]   (A = k rows, then x rows)
//   Stage 2 : P,Q = weighted complex product (in-place)
//   Stage 3 : y = P @ cosT + Q @ sinT
// TDM: one 3-D tensor_load_to_lds per K-step fetches all 4 basis tiles
// ([32 K][64 N][4 tables]) and hardware-pads each 64B row by 16B so the LDS
// image lands directly in the 80B-stride (20-bank) conflict-free layout.
// ---------------------------------------------------------------------------

typedef __bf16 bf16;
typedef __attribute__((ext_vector_type(16))) __bf16 v16bf;
typedef __attribute__((ext_vector_type(8)))  __bf16 v8bf;
typedef __attribute__((ext_vector_type(4)))  __bf16 v4bf;
typedef __attribute__((ext_vector_type(8)))  float  v8f;
typedef __attribute__((ext_vector_type(4)))  float  f32x4;
typedef __attribute__((ext_vector_type(4)))  unsigned int u32x4;
typedef __attribute__((ext_vector_type(8)))  int i32x8;
typedef __attribute__((ext_vector_type(4)))  int i32x4;

#define L_LEN  4096
#define NFREQ  512
#define H_DIM  768
#define NROWS  6144          // B*H = 8*768
#define LDSS   40            // LDS row stride in bf16: 80B = 20 banks (pad via TDM)
#define BTILE  (64 * LDSS)   // one 64-row basis tile in LDS (bf16 elems)

__device__ __forceinline__ v8f wmma_bf16(v16bf a, v16bf b, v8f c) {
  return __builtin_amdgcn_wmma_f32_16x16x32_bf16(false, a, false, b,
                                                 (short)0, c, false, false);
}

__device__ __forceinline__ v16bf cat8(v8bf a, v8bf b) {
  return __builtin_shufflevector(a, b, 0,1,2,3,4,5,6,7,8,9,10,11,12,13,14,15);
}

// A-matrix 16x32 fragment: lane L<16: M=L, K={0-7,16-23}; L>=16: K={8-15,24-31}
__device__ __forceinline__ v16bf fragA(const bf16* tile, int rowBase, int lane) {
  const bf16* p = tile + (size_t)(rowBase + (lane & 15)) * LDSS + ((lane >> 4) & 1) * 8;
  return cat8(*(const v8bf*)(p), *(const v8bf*)(p + 16));
}

// B-matrix 32x16 fragment from B^T rows: lane L<16: N=L,K={0-7,16-23}; L>=16: K={8-15,24-31}
__device__ __forceinline__ v16bf fragB(const bf16* tile, int rowBase, int lane) {
  const bf16* p = tile + (size_t)(rowBase + (lane & 15)) * LDSS + (lane >> 4) * 8;
  return cat8(*(const v8bf*)(p), *(const v8bf*)(p + 16));
}

__device__ __forceinline__ void split4(f32x4 v, v4bf& h, v4bf& l) {
  bf16 h0 = (bf16)v.x, h1 = (bf16)v.y, h2 = (bf16)v.z, h3 = (bf16)v.w;
  h = (v4bf){h0, h1, h2, h3};
  l = (v4bf){(bf16)(v.x - (float)h0), (bf16)(v.y - (float)h1),
             (bf16)(v.z - (float)h2), (bf16)(v.w - (float)h3)};
}

// ---------------------------------------------------------------------------
// TDM: 3-D tile [tile0 x][64 y][4 z] of bf16 from a [4][dim1][dim0] tensor.
// D# per CDNA5 ISA 8.3-8.6: count=1, type=2, data_size=2B,
// pad_enable, pad_interval=16 DWORDs (64B row), pad_amount=4 DWORDs (16B).
// ---------------------------------------------------------------------------
__device__ __forceinline__ void tdm_load_tiles(
    const bf16* gtile, unsigned ldsOff,
    unsigned dim0, unsigned long long stride0,
    unsigned dim1, unsigned long long stride1,
    unsigned tile0, unsigned tile1, unsigned ntab) {
  unsigned long long ga = (unsigned long long)gtile;
  u32x4 g0;
  g0.x = 1u;                                   // count=1 (user descriptor)
  g0.y = ldsOff;                               // LDS byte offset of dest
  g0.z = (unsigned)ga;                         // global_addr[31:0]
  g0.w = (unsigned)(ga >> 32) | 0x80000000u;   // global_addr[56:32] | type=2
  i32x8 g1;
  g1[0] = (1 << 16) | (1 << 20) | (3 << 22) | (3 << 25);   // 2B, pad 64B->+16B
  g1[1] = (int)((dim0 & 0xFFFFu) << 16);                   // tensor_dim0 lo
  g1[2] = (int)((dim0 >> 16) | ((dim1 & 0xFFFFu) << 16));  // dim0 hi | dim1 lo
  g1[3] = (int)((dim1 >> 16) | (tile0 << 16));             // dim1 hi | tile_dim0
  g1[4] = (int)(tile1 | (ntab << 16));                     // tile_dim1 | tile_dim2
  g1[5] = (int)(unsigned)(stride0 & 0xFFFFFFFFu);          // dim0_stride lo
  g1[6] = (int)((unsigned)(stride0 >> 32) |
                ((unsigned)(stride1 & 0xFFFFu) << 16));    // | dim1_stride lo
  g1[7] = (int)(unsigned)(stride1 >> 16);                  // dim1_stride hi
  i32x4 g2;
  g2[0] = (int)ntab;                           // tensor_dim2
  g2[1] = 0; g2[2] = 0; g2[3] = 0;             // dim3 / dim2_stride / tile3 unused
  i32x4 g3 = (i32x4)(0);
#if defined(__clang_major__) && __clang_major__ >= 23
  __builtin_amdgcn_tensor_load_to_lds(g0, g1, g2, g3, (i32x8)(0), 0);
#else
  __builtin_amdgcn_tensor_load_to_lds(g0, g1, g2, g3, 0);
#endif
}

// ---------------------------------------------------------------------------
// Basis tables: entry[i][j] = cos/sin(2*pi*(i*j mod 8192)/8192), bf16 hi/lo.
// Integer phase -> no large-argument trig reduction error.
// ---------------------------------------------------------------------------
__global__ __launch_bounds__(256) void build_tables(
    bf16* __restrict__ ch, bf16* __restrict__ cl,
    bf16* __restrict__ sh, bf16* __restrict__ sl,
    int cols, int total) {
  int idx = blockIdx.x * 256 + threadIdx.x;
  if (idx >= total) return;
  int i = idx / cols, j = idx - i * cols;
  int p = (i * j) & 8191;
  float th = (float)p * 7.6699039394282060e-4f;   // 2*pi/8192
  float c = cosf(th), s = sinf(th);
  bf16 chh = (bf16)c; ch[idx] = chh; cl[idx] = (bf16)(c - (float)chh);
  bf16 shh = (bf16)s; sh[idx] = shh; sl[idx] = (bf16)(s - (float)shh);
}

// ---------------------------------------------------------------------------
// Stage 1: Oc/Os[R,512] = A[R,4096] @ {cos,sin}[4096,512]
// 256 thr = 8 waves; 64x64 tile; wave = 2(M-half) x 4(N); 2 M-subtiles/wave
// -> 12 WMMAs per wave per 32-K step.
// ---------------------------------------------------------------------------
__global__ __launch_bounds__(256) void gemm_dft(
    const float* __restrict__ A,
    const bf16* __restrict__ Tbase,          // 4 contiguous [512][4096] tables
    float* __restrict__ Oc, float* __restrict__ Os) {
  __shared__ bf16 sAh[64 * LDSS], sAl[64 * LDSS];
  __shared__ bf16 sB[4 * BTILE];             // ch | cl | sh | sl

  const int t = threadIdx.x, lane = t & 31, wave = t >> 5;
  const int mh = (wave & 1) * 32, ni = wave >> 1;
  const int r0 = blockIdx.x * 64, c0 = blockIdx.y * 64;
  const int ar = t >> 3, ac = (t & 7) * 4;
  const unsigned ldsB = (unsigned)(unsigned long long)(void*)sB;

  v8f aC0 = {}, aC1 = {}, aS0 = {}, aS1 = {};

  for (int k0 = 0; k0 < L_LEN; k0 += 32) {
    __syncthreads();                               // prev compute done
    if (wave == 0)                                 // one TDM op per workgroup
      tdm_load_tiles(Tbase + (size_t)c0 * L_LEN + k0, ldsB,
                     L_LEN, L_LEN, NFREQ, (unsigned long long)NFREQ * L_LEN,
                     32, 64, 4);
#pragma unroll
    for (int rr = 0; rr < 64; rr += 32) {          // A tile: f32 -> bf16 hi/lo
      f32x4 av = *(const f32x4*)(A + (size_t)(r0 + ar + rr) * L_LEN + k0 + ac);
      v4bf h, l; split4(av, h, l);
      *(v4bf*)&sAh[(ar + rr) * LDSS + ac] = h;
      *(v4bf*)&sAl[(ar + rr) * LDSS + ac] = l;
    }
    if (wave == 0) __builtin_amdgcn_s_wait_tensorcnt(0);
    __syncthreads();                               // A + TDM tiles published

    v16bf Bch = fragB(sB + 0 * BTILE, ni * 16, lane);
    v16bf Bcl = fragB(sB + 1 * BTILE, ni * 16, lane);
    v16bf Bsh = fragB(sB + 2 * BTILE, ni * 16, lane);
    v16bf Bsl = fragB(sB + 3 * BTILE, ni * 16, lane);
    v16bf Ah0 = fragA(sAh, mh, lane),      Al0 = fragA(sAl, mh, lane);
    v16bf Ah1 = fragA(sAh, mh + 16, lane), Al1 = fragA(sAl, mh + 16, lane);
    // bf16x2 split: hi*hi + hi*lo + lo*hi, f32 accumulate
    aC0 = wmma_bf16(Ah0, Bch, aC0); aC0 = wmma_bf16(Ah0, Bcl, aC0); aC0 = wmma_bf16(Al0, Bch, aC0);
    aS0 = wmma_bf16(Ah0, Bsh, aS0); aS0 = wmma_bf16(Ah0, Bsl, aS0); aS0 = wmma_bf16(Al0, Bsh, aS0);
    aC1 = wmma_bf16(Ah1, Bch, aC1); aC1 = wmma_bf16(Ah1, Bcl, aC1); aC1 = wmma_bf16(Al1, Bch, aC1);
    aS1 = wmma_bf16(Ah1, Bsh, aS1); aS1 = wmma_bf16(Ah1, Bsl, aS1); aS1 = wmma_bf16(Al1, Bsh, aS1);
  }

  const int n = lane & 15, mq = (lane >> 4) * 8;
  const int gcol = c0 + ni * 16 + n;
#pragma unroll
  for (int i = 0; i < 8; ++i) {
    Oc[(size_t)(r0 + mh + mq + i) * NFREQ + gcol] = aC0[i];
    Os[(size_t)(r0 + mh + mq + i) * NFREQ + gcol] = aS0[i];
    Oc[(size_t)(r0 + mh + 16 + mq + i) * NFREQ + gcol] = aC1[i];
    Os[(size_t)(r0 + mh + 16 + mq + i) * NFREQ + gcol] = aS1[i];
  }
}

// ---------------------------------------------------------------------------
// Stage 2: P = a*(Xc*Kc - Xs*Ks), Q = a*(Xc*Ks + Xs*Kc); a = w_f/N.
// ---------------------------------------------------------------------------
__global__ __launch_bounds__(256) void modulate(
    float* __restrict__ Xc, float* __restrict__ Xs,
    const float* __restrict__ Kc, const float* __restrict__ Ks) {
  int idx = blockIdx.x * 256 + threadIdx.x;
  if (idx >= NROWS * NFREQ) return;
  int r = idx >> 9, f = idx & (NFREQ - 1);
  int h = r % H_DIM;
  float xc = Xc[idx], xs = Xs[idx];
  float kc = Kc[(size_t)h * NFREQ + f], ks = Ks[(size_t)h * NFREQ + f];
  float a = (f == 0) ? (1.0f / 8192.0f) : (2.0f / 8192.0f);
  Xc[idx] = a * (xc * kc - xs * ks);   // P
  Xs[idx] = a * (xc * ks + xs * kc);   // Q
}

// ---------------------------------------------------------------------------
// Stage 3: Y[6144,4096] = P @ cosT + Q @ sinT; tables B^T: [4][4096 n][512 f]
// 64x64 tile, 2 M-subtiles/wave -> 12 WMMAs per wave per 32-K step.
// ---------------------------------------------------------------------------
__global__ __launch_bounds__(256) void gemm_syn(
    const float* __restrict__ P, const float* __restrict__ Q,
    const bf16* __restrict__ Tbase,          // 4 contiguous [4096][512] tables
    float* __restrict__ Y) {
  __shared__ bf16 sPh[64 * LDSS], sPl[64 * LDSS];
  __shared__ bf16 sQh[64 * LDSS], sQl[64 * LDSS];
  __shared__ bf16 sB[4 * BTILE];

  const int t = threadIdx.x, lane = t & 31, wave = t >> 5;
  const int mh = (wave & 1) * 32, ni = wave >> 1;
  const int r0 = blockIdx.x * 64, c0 = blockIdx.y * 64;
  const int ar = t >> 3, ac = (t & 7) * 4;
  const unsigned ldsB = (unsigned)(unsigned long long)(void*)sB;

  v8f acc0 = {}, acc1 = {};

  for (int k0 = 0; k0 < NFREQ; k0 += 32) {
    __syncthreads();
    if (wave == 0)
      tdm_load_tiles(Tbase + (size_t)c0 * NFREQ + k0, ldsB,
                     NFREQ, NFREQ, L_LEN, (unsigned long long)NFREQ * L_LEN,
                     32, 64, 4);
#pragma unroll
    for (int rr = 0; rr < 64; rr += 32) {
      f32x4 pv = *(const f32x4*)(P + (size_t)(r0 + ar + rr) * NFREQ + k0 + ac);
      f32x4 qv = *(const f32x4*)(Q + (size_t)(r0 + ar + rr) * NFREQ + k0 + ac);
      v4bf h, l;
      split4(pv, h, l);
      *(v4bf*)&sPh[(ar + rr) * LDSS + ac] = h;
      *(v4bf*)&sPl[(ar + rr) * LDSS + ac] = l;
      split4(qv, h, l);
      *(v4bf*)&sQh[(ar + rr) * LDSS + ac] = h;
      *(v4bf*)&sQl[(ar + rr) * LDSS + ac] = l;
    }
    if (wave == 0) __builtin_amdgcn_s_wait_tensorcnt(0);
    __syncthreads();

    v16bf Bch = fragB(sB + 0 * BTILE, ni * 16, lane);
    v16bf Bcl = fragB(sB + 1 * BTILE, ni * 16, lane);
    v16bf Bsh = fragB(sB + 2 * BTILE, ni * 16, lane);
    v16bf Bsl = fragB(sB + 3 * BTILE, ni * 16, lane);
#pragma unroll
    for (int s = 0; s < 2; ++s) {
      v8f acc = s ? acc1 : acc0;
      v16bf Ph = fragA(sPh, mh + s * 16, lane), Pl = fragA(sPl, mh + s * 16, lane);
      v16bf Qh = fragA(sQh, mh + s * 16, lane), Ql = fragA(sQl, mh + s * 16, lane);
      acc = wmma_bf16(Ph, Bch, acc);
      acc = wmma_bf16(Ph, Bcl, acc);
      acc = wmma_bf16(Pl, Bch, acc);
      acc = wmma_bf16(Qh, Bsh, acc);
      acc = wmma_bf16(Qh, Bsl, acc);
      acc = wmma_bf16(Ql, Bsh, acc);
      if (s) acc1 = acc; else acc0 = acc;
    }
  }

  const int n = lane & 15, mq = (lane >> 4) * 8;
  const int gcol = c0 + ni * 16 + n;
#pragma unroll
  for (int i = 0; i < 8; ++i) {
    Y[(size_t)(r0 + mh + mq + i) * L_LEN + gcol] = acc0[i];
    Y[(size_t)(r0 + mh + 16 + mq + i) * L_LEN + gcol] = acc1[i];
  }
}

// ---------------------------------------------------------------------------
extern "C" void kernel_launch(void* const* d_in, const int* in_sizes, int n_in,
                              void* d_out, int out_size, void* d_ws, size_t ws_size,
                              hipStream_t stream) {
  (void)in_sizes; (void)n_in; (void)out_size; (void)ws_size;
  const float* x = (const float*)d_in[0];   // [8,768,4096] f32
  const float* k = (const float*)d_in[1];   // [768,4096]   f32
  float* y = (float*)d_out;                 // [8,768,4096] f32

  // Workspace (~59 MiB): 8 basis tables (contiguous per stage), K and X spectra
  char* ws = (char*)d_ws;
  const size_t TBL = (size_t)NFREQ * L_LEN * sizeof(bf16);   // 4 MiB per table
  bf16* T1 = (bf16*)(ws + 0 * TBL);   // [4][512 f][4096 m]: ch,cl,sh,sl
  bf16* T3 = (bf16*)(ws + 4 * TBL);   // [4][4096 n][512 f]: ch,cl,sh,sl
  size_t off = 8 * TBL;
  float* Kc = (float*)(ws + off); off += (size_t)H_DIM * NFREQ * sizeof(float);
  float* Ks = (float*)(ws + off); off += (size_t)H_DIM * NFREQ * sizeof(float);
  float* Xc = (float*)(ws + off); off += (size_t)NROWS * NFREQ * sizeof(float);
  float* Xs = (float*)(ws + off);

  const int tblN = NFREQ * L_LEN;     // 2M entries per table
  const size_t TE = (size_t)tblN;     // element stride between tables
  build_tables<<<(tblN + 255) / 256, 256, 0, stream>>>(
      T1, T1 + TE, T1 + 2 * TE, T1 + 3 * TE, L_LEN, tblN);
  build_tables<<<(tblN + 255) / 256, 256, 0, stream>>>(
      T3, T3 + TE, T3 + 2 * TE, T3 + 3 * TE, NFREQ, tblN);

  // Kernel spectrum (768 rows), then signal spectrum (6144 rows)
  gemm_dft<<<dim3(H_DIM / 64, NFREQ / 64), 256, 0, stream>>>(k, T1, Kc, Ks);
  gemm_dft<<<dim3(NROWS / 64, NFREQ / 64), 256, 0, stream>>>(x, T1, Xc, Xs);

  const int modN = NROWS * NFREQ;
  modulate<<<(modN + 255) / 256, 256, 0, stream>>>(Xc, Xs, Kc, Ks);

  gemm_syn<<<dim3(NROWS / 64, L_LEN / 64), 256, 0, stream>>>(Xc, Xs, T3, y);
}
